// Decoder_84894323572745
// MI455X (gfx1250) — matrix operand order repaired and checked by reading, not compiled, not run on hardware
//
#include <hip/hip_runtime.h>

#define DEV __device__ __forceinline__

typedef __attribute__((ext_vector_type(16))) __bf16 v16bf;
typedef __attribute__((ext_vector_type(8)))  float  v8f;

constexpr int Bc = 32, Sc = 32, Fc = 10, Lc = 64, Dc = 512, Ec = 512;
constexpr int UNITS = 512, Acst = 512, READOUT = 1024, VOCABc = 32000, Tc = 24;
constexpr int IN_DIM = Ec + Dc + Ec;        // 1536
constexpr int XH     = IN_DIM + UNITS;      // 2048 (x | h)
constexpr int RIN    = UNITS + IN_DIM + Ec; // 2560 (h2 | x | qvec)
constexpr int FL     = Fc * Lc;             // 640

// float -> bf16, round-to-nearest-even
DEV unsigned short f2bf(float f) {
    unsigned int u = __float_as_uint(f);
    unsigned int r = u + 0x7FFFu + ((u >> 16) & 1u);
    return (unsigned short)(r >> 16);
}

// A-operand k mapping (16-bit A 16x32 layout): element i, lane-half `half`
DEV int kmapA(int half, int i) {
    int v = i >> 1, s = i & 1;
    return (v < 4) ? (half * 8 + 2 * v + s) : (16 + half * 8 + 2 * (v - 4) + s);
}

// ---------------------------------------------------------------------------
// Pack row-major fp32 A[M,K] into WMMA-A bf16 fragments.
// dst[(mTile*kChunks + kc)*512 + lane*16 + i];  m = mTile*16 + (lane&15)
// grid = mTiles*kChunks, block = 512
// ---------------------------------------------------------------------------
__global__ __launch_bounds__(512)
void pack_a(const float* __restrict__ A, int K, unsigned short* __restrict__ dst,
            int kChunks)
{
    int tid  = threadIdx.x;
    int lane = tid >> 4;
    int i    = tid & 15;
    int blk  = blockIdx.x;            // mTile*kChunks + kc
    int kc   = blk % kChunks;
    int mT   = blk / kChunks;
    int m    = mT * 16 + (lane & 15);
    int k    = kc * 32 + kmapA(lane >> 4, i);
    dst[(size_t)blk * 512 + tid] = f2bf(A[(size_t)m * K + k]);
}

// ---------------------------------------------------------------------------
// Pack (up to 3 stacked sources) row-major fp32 [K,N] weight into WMMA-B
// bf16 fragments: lane l: n = nTile*16 + (l&15); k = kc*32 + kmapA(l>>4, i)
// dst[(nTile*kChunks + kc)*512 + lane*16 + i]
// ---------------------------------------------------------------------------
__global__ __launch_bounds__(512)
void pack_b(const float* __restrict__ s0, int k0,
            const float* __restrict__ s1, int k1,
            const float* __restrict__ s2, int k2,
            int N, unsigned short* __restrict__ dst, int kChunks)
{
    int tid  = threadIdx.x;
    int lane = tid >> 4;
    int i    = tid & 15;
    int blk  = blockIdx.x;            // nt*kChunks + kc
    int kc   = blk % kChunks;
    int nt   = blk / kChunks;
    int n    = nt * 16 + (lane & 15);
    int k    = kc * 32 + kmapA(lane >> 4, i);
    float v;
    if (k < k0)            v = s0[(size_t)k * N + n];
    else if (k < k0 + k1)  v = s1[(size_t)(k - k0) * N + n];
    else                   v = s2[(size_t)(k - k0 - k1) * N + n];
    dst[(size_t)blk * 512 + tid] = f2bf(v);
}

// ---------------------------------------------------------------------------
// Fully-packed, ping-pong pipelined GEMM: C[M,N] = Apacked x Bpacked + bias.
// One wave per mTile x NT nTiles. Two independent fragment buffers are
// alternately consumed by WMMAs and refilled in place for k-chunk +2, so no
// buffer-rotation register copies exist and loads overlap the matrix pipe.
// Requires kChunks even and >= 4. grid = (nTiles/NT, mTiles), block = 32.
// ---------------------------------------------------------------------------
template <int NT>
__global__ __launch_bounds__(32)
void gemm_pp(const unsigned short* __restrict__ Ap,
             const unsigned short* __restrict__ Bp,
             const float* __restrict__ bias, float* __restrict__ C,
             int N, int kChunks)
{
    int lane = threadIdx.x;
    int half = lane >> 4;
    int l15  = lane & 15;
    int mT   = blockIdx.y;
    int nT0  = blockIdx.x * NT;

    v8f acc[NT];
    #pragma unroll
    for (int nt = 0; nt < NT; ++nt) acc[nt] = v8f{0.f,0.f,0.f,0.f,0.f,0.f,0.f,0.f};

    const v16bf* ap = (const v16bf*)Ap + (size_t)mT * kChunks * 32 + lane;
    const v16bf* bp = (const v16bf*)Bp + (size_t)nT0 * kChunks * 32 + lane;
    const size_t bts = (size_t)kChunks * 32;   // v16bf per nTile

    // prologue: fill both buffers (k-chunks 0 and 1)
    v16bf a0 = ap[0];
    v16bf a1 = ap[32];
    v16bf b0[NT], b1[NT];
    #pragma unroll
    for (int nt = 0; nt < NT; ++nt) {
        b0[nt] = bp[(size_t)nt * bts];
        b1[nt] = bp[(size_t)nt * bts + 32];
    }

    for (int kc = 0; kc < kChunks - 2; kc += 2) {
        #pragma unroll
        for (int nt = 0; nt < NT; ++nt)
            acc[nt] = __builtin_amdgcn_wmma_f32_16x16x32_bf16(
                          false, a0, false, b0[nt], (short)0, acc[nt], false, false);
        a0 = ap[(size_t)(kc + 2) * 32];
        #pragma unroll
        for (int nt = 0; nt < NT; ++nt)
            b0[nt] = bp[(size_t)nt * bts + (size_t)(kc + 2) * 32];

        #pragma unroll
        for (int nt = 0; nt < NT; ++nt)
            acc[nt] = __builtin_amdgcn_wmma_f32_16x16x32_bf16(
                          false, a1, false, b1[nt], (short)0, acc[nt], false, false);
        a1 = ap[(size_t)(kc + 3) * 32];
        #pragma unroll
        for (int nt = 0; nt < NT; ++nt)
            b1[nt] = bp[(size_t)nt * bts + (size_t)(kc + 3) * 32];
    }
    // epilogue: consume the final two chunks
    #pragma unroll
    for (int nt = 0; nt < NT; ++nt)
        acc[nt] = __builtin_amdgcn_wmma_f32_16x16x32_bf16(
                      false, a0, false, b0[nt], (short)0, acc[nt], false, false);
    #pragma unroll
    for (int nt = 0; nt < NT; ++nt)
        acc[nt] = __builtin_amdgcn_wmma_f32_16x16x32_bf16(
                      false, a1, false, b1[nt], (short)0, acc[nt], false, false);

    #pragma unroll
    for (int nt = 0; nt < NT; ++nt) {
        int n = (nT0 + nt) * 16 + l15;
        float bv = bias ? bias[n] : 0.f;
        #pragma unroll
        for (int r = 0; r < 8; ++r) {
            int row = mT * 16 + half * 8 + r;   // lanes16-31 -> M+8
            C[(size_t)row * N + n] = acc[nt][r] + bv;
        }
    }
}

// ---------------------------------------------------------------------------
// Question attention: scores over S, softmax, weighted sum of base embeddings.
// grid = B, block = (32,8)
// ---------------------------------------------------------------------------
__global__ __launch_bounds__(256)
void qattn(const float* __restrict__ qproj, const float* __restrict__ hWq2,
           const float* __restrict__ vq, const float* __restrict__ baseq,
           float* __restrict__ qvec)
{
    int b = blockIdx.x;
    int lx = threadIdx.x, ly = threadIdx.y;
    __shared__ float sc[Sc];
    __shared__ float wt[Sc];

    const float* hw = hWq2 + (size_t)b * Acst;
    #pragma unroll
    for (int q = 0; q < 4; ++q) {
        int s = ly + 8 * q;
        const float* qp = qproj + ((size_t)b * Sc + s) * Acst;
        float acc = 0.f;
        for (int a = lx; a < Acst; a += 32)
            acc += vq[a] * tanhf(qp[a] + hw[a]);
        #pragma unroll
        for (int off = 16; off > 0; off >>= 1) acc += __shfl_xor(acc, off, 32);
        if (lx == 0) sc[s] = acc;
    }
    __syncthreads();
    if (ly == 0) {
        float v = sc[lx];
        float mx = v;
        #pragma unroll
        for (int off = 16; off > 0; off >>= 1) mx = fmaxf(mx, __shfl_xor(mx, off, 32));
        float e = expf(v - mx);
        float z = e;
        #pragma unroll
        for (int off = 16; off > 0; off >>= 1) z += __shfl_xor(z, off, 32);
        wt[lx] = e / z;
    }
    __syncthreads();
    int tid = ly * 32 + lx;
    for (int e0 = tid; e0 < Ec; e0 += 256) {
        float acc = 0.f;
        #pragma unroll 4
        for (int s = 0; s < Sc; ++s)
            acc += wt[s] * baseq[((size_t)b * Sc + s) * Ec + e0];
        qvec[(size_t)b * Ec + e0] = acc;
    }
}

// Fact scores: fs[b,j] = vf . tanh(f_proj[b,j,:] + hWf2[b,:])
// grid = (FL/8, B), block = (32,8)
__global__ __launch_bounds__(256)
void fscore(const float* __restrict__ fproj, const float* __restrict__ hWf2,
            const float* __restrict__ vf, float* __restrict__ fs)
{
    int b = blockIdx.y;
    int j = blockIdx.x * 8 + threadIdx.y;
    int lx = threadIdx.x;
    const float* fp = fproj + ((size_t)b * FL + j) * Acst;
    const float* hw = hWf2 + (size_t)b * Acst;
    float acc = 0.f;
    for (int a = lx; a < Acst; a += 32) acc += vf[a] * tanhf(fp[a] + hw[a]);
    #pragma unroll
    for (int off = 16; off > 0; off >>= 1) acc += __shfl_xor(acc, off, 32);
    if (lx == 0) fs[(size_t)b * FL + j] = acc;
}

// Exact top-64 (rank = #strictly-greater + equal-with-smaller-index, matching
// lax.top_k stable ordering), softmax over selected, weighted fact sum.
// grid = B, block = 256
__global__ __launch_bounds__(256)
void topk_fvec(const float* __restrict__ fs, const float* __restrict__ facts,
               float* __restrict__ fvec)
{
    int b = blockIdx.x, tid = threadIdx.x;
    __shared__ float s[FL];
    __shared__ float selS[Lc];
    __shared__ int   selI[Lc];
    __shared__ float w[Lc];
    __shared__ float zsh;

    for (int j = tid; j < FL; j += 256) s[j] = fs[(size_t)b * FL + j];
    __syncthreads();
    for (int j = tid; j < FL; j += 256) {
        float sj = s[j];
        int rank = 0;
        for (int jj = 0; jj < FL; ++jj) {
            float v = s[jj];
            rank += (v > sj) || (v == sj && jj < j);
        }
        if (rank < Lc) { selS[rank] = sj; selI[rank] = j; }
    }
    __syncthreads();
    if (tid < Lc) w[tid] = expf(selS[tid] - selS[0]);   // rank0 == max
    __syncthreads();
    if (tid == 0) { float z = 0.f; for (int r = 0; r < Lc; ++r) z += w[r]; zsh = z; }
    __syncthreads();
    if (tid < Lc) w[tid] /= zsh;
    __syncthreads();
    for (int d = tid; d < Dc; d += 256) {
        float acc = 0.f;
        for (int r = 0; r < Lc; ++r)
            acc += w[r] * facts[((size_t)b * FL + selI[r]) * Dc + d];
        fvec[(size_t)b * Dc + d] = acc;
    }
}

// xh[b] = [prev | fvec | qvec | h] ; prev = (t==0)?0:embedding[target[b,t-1]]
__global__ __launch_bounds__(512)
void assemble_xh(float* __restrict__ xh, const float* __restrict__ h,
                 const float* __restrict__ fvec, const float* __restrict__ qvec,
                 const float* __restrict__ emb, const int* __restrict__ target, int t)
{
    int b = blockIdx.x, u = threadIdx.x;
    float prev = 0.f;
    if (t > 0) {
        int tok = target[b * Tc + (t - 1)];
        prev = emb[(size_t)tok * Ec + u];
    }
    float* row = xh + (size_t)b * XH;
    row[u]            = prev;
    row[Ec + u]       = fvec[(size_t)b * Dc + u];
    row[Ec + Dc + u]  = qvec[(size_t)b * Ec + u];
    row[IN_DIM + u]   = h[(size_t)b * UNITS + u];
}

// LSTM pointwise (keras gate order i,f,g,o); updates h,c in place
__global__ __launch_bounds__(512)
void lstm_pw(const float* __restrict__ zg, float* __restrict__ h, float* __restrict__ c)
{
    int b = blockIdx.x, u = threadIdx.x;
    const float* z = zg + (size_t)b * (4 * UNITS);
    float gi = z[u], gf = z[UNITS + u], gg = z[2 * UNITS + u], go = z[3 * UNITS + u];
    float si = 1.f / (1.f + expf(-gi));
    float sf = 1.f / (1.f + expf(-gf));
    float so = 1.f / (1.f + expf(-go));
    float c2 = sf * c[(size_t)b * UNITS + u] + si * tanhf(gg);
    float h2 = so * tanhf(c2);
    c[(size_t)b * UNITS + u] = c2;
    h[(size_t)b * UNITS + u] = h2;
}

// rin[b] = [h2 | x | qvec]
__global__ __launch_bounds__(256)
void assemble_rin(float* __restrict__ rin, const float* __restrict__ h,
                  const float* __restrict__ xh, const float* __restrict__ qvec)
{
    int b = blockIdx.x, tid = threadIdx.x;
    float* row = rin + (size_t)b * RIN;
    const float* x = xh + (size_t)b * XH;
    for (int u = tid; u < RIN; u += 256) {
        float v;
        if (u < UNITS)               v = h[(size_t)b * UNITS + u];
        else if (u < UNITS + IN_DIM) v = x[u - UNITS];
        else                         v = qvec[(size_t)b * Ec + (u - UNITS - IN_DIM)];
        row[u] = v;
    }
}

__global__ __launch_bounds__(512)
void maxout_k(const float* __restrict__ r, float* __restrict__ m)
{
    int b = blockIdx.x, u = threadIdx.x;
    const float* rr = r + (size_t)b * READOUT;
    m[(size_t)b * (READOUT / 2) + u] = fmaxf(rr[2 * u], rr[2 * u + 1]);
}

__global__ __launch_bounds__(256)
void combine_bias(const float* __restrict__ a, const float* __restrict__ b2,
                  const float* __restrict__ c2, float* __restrict__ o, int n)
{
    int i = blockIdx.x * 256 + threadIdx.x;
    if (i < n) o[i] = a[i] + b2[i] + c2[i];
}

__global__ __launch_bounds__(256)
void zero_f(float* __restrict__ p, int n)
{
    int i = blockIdx.x * 256 + threadIdx.x;
    if (i < n) p[i] = 0.f;
}

// ---------------------------------------------------------------------------
extern "C" void kernel_launch(void* const* d_in, const int* in_sizes, int n_in,
                              void* d_out, int out_size, void* d_ws, size_t ws_size,
                              hipStream_t stream)
{
    const float* baseq   = (const float*)d_in[0];
    const float* facts   = (const float*)d_in[1];
    const int*   target  = (const int*)d_in[2];
    // d_in[3] = nb_steps (constant 24)
    const float* emb     = (const float*)d_in[4];
    const float* lstm_k  = (const float*)d_in[5];
    const float* lstm_rk = (const float*)d_in[6];
    const float* lstm_b  = (const float*)d_in[7];
    const float* Wq1 = (const float*)d_in[8];
    const float* Wq2 = (const float*)d_in[9];
    const float* vq  = (const float*)d_in[10];
    const float* Wf1 = (const float*)d_in[11];
    const float* Wf2 = (const float*)d_in[12];
    const float* vf  = (const float*)d_in[13];
    const float* Wr  = (const float*)d_in[14];
    const float* br  = (const float*)d_in[15];
    const float* Ur  = (const float*)d_in[16];
    const float* bur = (const float*)d_in[17];
    const float* Vr  = (const float*)d_in[18];
    const float* bvr = (const float*)d_in[19];
    const float* Wy  = (const float*)d_in[20];
    const float* by  = (const float*)d_in[21];
    float* out = (float*)d_out;

    char* ws = (char*)d_ws;
    size_t off = 0;
    auto alloc = [&](size_t bytes) -> char* {
        char* p = ws + off;
        off += (bytes + 255) & ~(size_t)255;
        return p;
    };

    // packed weights (B operands)
    unsigned short* pWq1  = (unsigned short*)alloc((size_t)512 * 512 * 2);
    unsigned short* pWq2  = (unsigned short*)alloc((size_t)512 * 512 * 2);
    unsigned short* pWf1  = (unsigned short*)alloc((size_t)512 * 512 * 2);
    unsigned short* pWf2  = (unsigned short*)alloc((size_t)512 * 512 * 2);
    unsigned short* pLSTM = (unsigned short*)alloc((size_t)2048 * 2048 * 2);
    unsigned short* pRead = (unsigned short*)alloc((size_t)2560 * 1024 * 2);
    unsigned short* pWy   = (unsigned short*)alloc((size_t)512 * VOCABc * 2);
    // packed activations (A operands)
    unsigned short* aBase = (unsigned short*)alloc((size_t)Bc * Sc * Ec * 2);
    unsigned short* aFact = (unsigned short*)alloc((size_t)Bc * FL * Dc * 2);
    unsigned short* aH    = (unsigned short*)alloc((size_t)Bc * UNITS * 2);
    unsigned short* aXH   = (unsigned short*)alloc((size_t)Bc * XH * 2);
    unsigned short* aRIN  = (unsigned short*)alloc((size_t)Bc * RIN * 2);
    unsigned short* aM    = (unsigned short*)alloc((size_t)Bc * (READOUT / 2) * 2);
    // fp32 scratch
    float* q_proj = (float*)alloc((size_t)Bc * Sc * Acst * 4);
    float* f_proj = (float*)alloc((size_t)Bc * FL * Acst * 4);
    float* hWq2b  = (float*)alloc((size_t)Bc * Acst * 4);
    float* hWf2b  = (float*)alloc((size_t)Bc * Acst * 4);
    float* fsb    = (float*)alloc((size_t)Bc * FL * 4);
    float* qvecb  = (float*)alloc((size_t)Bc * Ec * 4);
    float* fvecb  = (float*)alloc((size_t)Bc * Dc * 4);
    float* hb     = (float*)alloc((size_t)Bc * UNITS * 4);
    float* cb     = (float*)alloc((size_t)Bc * UNITS * 4);
    float* xhb    = (float*)alloc((size_t)Bc * XH * 4);
    float* zgb    = (float*)alloc((size_t)Bc * 4 * UNITS * 4);
    float* rinb   = (float*)alloc((size_t)Bc * RIN * 4);
    float* rb     = (float*)alloc((size_t)Bc * READOUT * 4);
    float* mb     = (float*)alloc((size_t)Bc * (READOUT / 2) * 4);
    float* brc    = (float*)alloc((size_t)READOUT * 4);
    (void)ws_size; (void)in_sizes; (void)n_in; (void)out_size;

    // ---- weight packing (deterministic, every call) ----
    pack_b<<<32 * 16, 512, 0, stream>>>(Wq1, 512, nullptr, 0, nullptr, 0, 512, pWq1, 16);
    pack_b<<<32 * 16, 512, 0, stream>>>(Wq2, 512, nullptr, 0, nullptr, 0, 512, pWq2, 16);
    pack_b<<<32 * 16, 512, 0, stream>>>(Wf1, 512, nullptr, 0, nullptr, 0, 512, pWf1, 16);
    pack_b<<<32 * 16, 512, 0, stream>>>(Wf2, 512, nullptr, 0, nullptr, 0, 512, pWf2, 16);
    pack_b<<<128 * 64, 512, 0, stream>>>(lstm_k, IN_DIM, lstm_rk, UNITS, nullptr, 0,
                                         4 * UNITS, pLSTM, 64);
    pack_b<<<64 * 80, 512, 0, stream>>>(Wr, UNITS, Ur, IN_DIM, Vr, Ec,
                                        READOUT, pRead, 80);
    pack_b<<<2000 * 16, 512, 0, stream>>>(Wy, 512, nullptr, 0, nullptr, 0,
                                          VOCABc, pWy, 16);
    combine_bias<<<4, 256, 0, stream>>>(br, bur, bvr, brc, READOUT);
    zero_f<<<(Bc * UNITS + 255) / 256, 256, 0, stream>>>(hb, Bc * UNITS);
    zero_f<<<(Bc * UNITS + 255) / 256, 256, 0, stream>>>(cb, Bc * UNITS);

    // ---- hoisted encoder projections (pack A once, WMMA GEMM) ----
    pack_a<<<64 * 16, 512, 0, stream>>>(baseq, Ec, aBase, 16);
    pack_a<<<1280 * 16, 512, 0, stream>>>(facts, Dc, aFact, 16);
    gemm_pp<4><<<dim3(Acst / 64, (Bc * Sc) / 16), 32, 0, stream>>>(
        aBase, pWq1, nullptr, q_proj, Acst, 16);
    gemm_pp<4><<<dim3(Acst / 64, (Bc * FL) / 16), 32, 0, stream>>>(
        aFact, pWf1, nullptr, f_proj, Acst, 16);

    // ---- recurrence ----
    for (int t = 0; t < Tc; ++t) {
        pack_a<<<2 * 16, 512, 0, stream>>>(hb, UNITS, aH, 16);
        gemm_pp<4><<<dim3(Acst / 64, Bc / 16), 32, 0, stream>>>(
            aH, pWq2, nullptr, hWq2b, Acst, 16);
        gemm_pp<4><<<dim3(Acst / 64, Bc / 16), 32, 0, stream>>>(
            aH, pWf2, nullptr, hWf2b, Acst, 16);
        qattn<<<Bc, dim3(32, 8), 0, stream>>>(q_proj, hWq2b, vq, baseq, qvecb);
        fscore<<<dim3(FL / 8, Bc), dim3(32, 8), 0, stream>>>(f_proj, hWf2b, vf, fsb);
        topk_fvec<<<Bc, 256, 0, stream>>>(fsb, facts, fvecb);
        assemble_xh<<<Bc, 512, 0, stream>>>(xhb, hb, fvecb, qvecb, emb, target, t);
        pack_a<<<2 * 64, 512, 0, stream>>>(xhb, XH, aXH, 64);
        gemm_pp<4><<<dim3((4 * UNITS) / 64, Bc / 16), 32, 0, stream>>>(
            aXH, pLSTM, lstm_b, zgb, 4 * UNITS, 64);
        lstm_pw<<<Bc, 512, 0, stream>>>(zgb, hb, cb);
        assemble_rin<<<Bc, 256, 0, stream>>>(rinb, hb, xhb, qvecb);
        pack_a<<<2 * 80, 512, 0, stream>>>(rinb, RIN, aRIN, 80);
        gemm_pp<4><<<dim3(READOUT / 64, Bc / 16), 32, 0, stream>>>(
            aRIN, pRead, brc, rb, READOUT, 80);
        maxout_k<<<Bc, 512, 0, stream>>>(rb, mb);
        pack_a<<<2 * 16, 512, 0, stream>>>(mb, READOUT / 2, aM, 16);
        gemm_pp<4><<<dim3(VOCABc / 64, Bc / 16), 32, 0, stream>>>(
            aM, pWy, by, out + (size_t)t * Bc * VOCABc, VOCABc, 16);
    }
}
